// SDARMoeSparseMoeBlock_84499186582099
// MI455X (gfx1250) — compile-verified, matrix operand order checked by reading, not run on hardware
//
#include <hip/hip_runtime.h>
#include <hip/hip_bf16.h>
#include <cstdint>
#include <cstddef>

// ---------------- CDNA5 (gfx1250) MoE SwiGLU block, bf16 WMMA ----------------
// D = A x B + C with v_wmma_f32_16x16x32_bf16 (wave32).
// hs tile staged with GLOBAL_LOAD_ASYNC_TO_LDS_B128 (ASYNCcnt).

typedef __attribute__((ext_vector_type(16))) __bf16 v16bf;
typedef __attribute__((ext_vector_type(8)))  float  v8f;

#define T_DIM 1024
#define H_DIM 1024
#define I_DIM 1408
#define E_DIM 16
#define TT    32                 // tokens per expert workgroup
#define HS_PITCH (H_DIM + 8)     // bf16 elements, pad to spread LDS banks
#define ACT_PITCH (I_DIM + 8)
#define MAX_SLOTS (T_DIM * 2)

// ---- fragment helpers -------------------------------------------------------
// A fragment (16x32 bf16, M rows in lanes): lane l holds row M=(l&15),
// K elems {8*hi..8*hi+7} then {16+8*hi..16+8*hi+7}, hi = l>>4.
// Source here is fp32 row-major; convert to bf16 in registers.
__device__ __forceinline__ v16bf load_a_f32(const float* __restrict__ p) {
  const float4* q = reinterpret_cast<const float4*>(p);
  float4 f0 = q[0], f1 = q[1];      // k = +0..7   (after caller adds 8*hi)
  float4 f2 = q[4], f3 = q[5];      // k = +16..23
  v16bf r;
  r[0]=(__bf16)f0.x; r[1]=(__bf16)f0.y; r[2]=(__bf16)f0.z; r[3]=(__bf16)f0.w;
  r[4]=(__bf16)f1.x; r[5]=(__bf16)f1.y; r[6]=(__bf16)f1.z; r[7]=(__bf16)f1.w;
  r[8]=(__bf16)f2.x; r[9]=(__bf16)f2.y; r[10]=(__bf16)f2.z; r[11]=(__bf16)f2.w;
  r[12]=(__bf16)f3.x; r[13]=(__bf16)f3.y; r[14]=(__bf16)f3.z; r[15]=(__bf16)f3.w;
  return r;
}

// B fragment (32x16 bf16, N cols in lanes): lane l holds col N=(l&15),
// same K element sequence as A. Source is bf16 in LDS, K-contiguous.
__device__ __forceinline__ v16bf load_b_lds(const __bf16* p) {
  union { v16bf v; uint4 q[2]; } u;
  u.q[0] = *reinterpret_cast<const uint4*>(p);        // k +0..7
  u.q[1] = *reinterpret_cast<const uint4*>(p + 16);   // k +16..23
  return u.v;
}

__device__ __forceinline__ v8f wmma_bf16(v16bf a, v16bf b, v8f c) {
  return __builtin_amdgcn_wmma_f32_16x16x32_bf16(false, a, false, b,
                                                 (short)0, c, false, false);
}

// ---- kernel 1: convert hidden_states to bf16, zero counters -----------------
__global__ __launch_bounds__(256) void k_init(const float* __restrict__ hs,
                                              __bf16* __restrict__ hsb,
                                              int* __restrict__ counts,
                                              int* __restrict__ cursors) {
  int idx = blockIdx.x * blockDim.x + threadIdx.x;
  int stride = gridDim.x * blockDim.x;
  for (int i = idx; i < T_DIM * H_DIM; i += stride)
    hsb[i] = (__bf16)hs[i];
  if (blockIdx.x == 0 && threadIdx.x < 2 * E_DIM) {
    if (threadIdx.x < E_DIM) counts[threadIdx.x] = 0;
    else                     cursors[threadIdx.x - E_DIM] = 0;
  }
}

// ---- kernel 2: router (fp32 logits, softmax top-2, renormalize) -------------
__global__ __launch_bounds__(256) void k_router(const float* __restrict__ hs,
                                                const float* __restrict__ gw,
                                                int* __restrict__ counts,
                                                int* __restrict__ top_idx,
                                                float* __restrict__ top_w) {
  __shared__ float logits[16][E_DIM + 1];
  int tid = threadIdx.x;
  int tl = tid >> 4, e = tid & 15;
  int t = blockIdx.x * 16 + tl;
  const float4* hrow = reinterpret_cast<const float4*>(hs + (size_t)t * H_DIM);
  const float4* grow = reinterpret_cast<const float4*>(gw + (size_t)e * H_DIM);
  float acc = 0.f;
  for (int h = 0; h < H_DIM / 4; ++h) {
    float4 a = hrow[h], b = grow[h];
    acc += a.x * b.x + a.y * b.y + a.z * b.z + a.w * b.w;
  }
  logits[tl][e] = acc;
  __syncthreads();
  if (tid < 16) {
    int tt = blockIdx.x * 16 + tid;
    float l[E_DIM];
    float m = -1e30f;
    for (int i = 0; i < E_DIM; ++i) { l[i] = logits[tid][i]; m = fmaxf(m, l[i]); }
    int i0 = 0; float v0 = -1e30f;
    for (int i = 0; i < E_DIM; ++i) if (l[i] > v0) { v0 = l[i]; i0 = i; }
    int i1 = (i0 == 0) ? 1 : 0; float v1 = -1e30f;
    for (int i = 0; i < E_DIM; ++i) if (i != i0 && l[i] > v1) { v1 = l[i]; i1 = i; }
    float p0 = __expf(v0 - m), p1 = __expf(v1 - m);
    float inv = 1.f / (p0 + p1);
    top_idx[tt * 2 + 0] = i0; top_idx[tt * 2 + 1] = i1;
    top_w[tt * 2 + 0] = p0 * inv; top_w[tt * 2 + 1] = p1 * inv;
    atomicAdd(&counts[i0], 1); atomicAdd(&counts[i1], 1);
  }
}

// ---- kernel 3: exclusive prefix sum over expert counts ----------------------
__global__ void k_scan(const int* __restrict__ counts, int* __restrict__ offsets) {
  if (threadIdx.x == 0 && blockIdx.x == 0) {
    int acc = 0;
    for (int e = 0; e < E_DIM; ++e) { offsets[e] = acc; acc += counts[e]; }
  }
}

// ---- kernel 4: build compact expert->token lists ----------------------------
__global__ __launch_bounds__(256) void k_fill(const int* __restrict__ top_idx,
                                              const int* __restrict__ offsets,
                                              int* __restrict__ cursors,
                                              int* __restrict__ list_tok,
                                              int* __restrict__ slot_of) {
  int t = blockIdx.x * blockDim.x + threadIdx.x;
  if (t < T_DIM) {
    for (int k = 0; k < 2; ++k) {
      int e = top_idx[t * 2 + k];
      int j = atomicAdd(&cursors[e], 1);
      int slot = offsets[e] + j;
      list_tok[slot] = t;
      slot_of[t * 2 + k] = slot;
    }
  }
}

// ---- kernel 5: fused SwiGLU expert GEMMs (bf16 WMMA, fp32 accum) ------------
// grid: (chunk, expert). Block = 256 threads = 8 wave32.
__global__ __launch_bounds__(256) void k_expert(const float* __restrict__ wg,
                                                const float* __restrict__ wu,
                                                const float* __restrict__ wd,
                                                const __bf16* __restrict__ hsb,
                                                const int* __restrict__ counts,
                                                const int* __restrict__ offsets,
                                                const int* __restrict__ list_tok,
                                                float* __restrict__ part) {
  __shared__ __align__(16) __bf16 hs_lds[TT * HS_PITCH];    // 64.5 KB
  __shared__ __align__(16) __bf16 act_lds[TT * ACT_PITCH];  // 88.5 KB
  __shared__ int tok_lds[TT];

  const int e = blockIdx.y;
  const int chunk = blockIdx.x;
  const int n_e = counts[e];
  const int t0 = chunk * TT;
  if (t0 >= n_e) return;
  const int nvalid = min(TT, n_e - t0);
  const int base_slot = offsets[e] + t0;
  const int tid = threadIdx.x;

  if (tid < TT)
    tok_lds[tid] = list_tok[base_slot + ((tid < nvalid) ? tid : 0)];
  __syncthreads();

  // Stage 32 bf16 token rows into LDS via async memory->LDS DMA (no VGPR
  // round-trip); GVS mode: vdst = per-lane LDS byte address, vaddr = 32-bit
  // byte offset, saddr = uniform 64-bit base. Tracked by ASYNCcnt.
  {
    int row = tid >> 3;                 // 0..31
    int seg = (tid & 7) * 128;          // 8 threads per row
    unsigned goff = (unsigned)(((size_t)tok_lds[row] * H_DIM + seg) *
                               sizeof(__bf16));
    unsigned ldsa = (unsigned)(uintptr_t)(hs_lds + row * HS_PITCH + seg);
#pragma unroll
    for (int j = 0; j < 8; ++j) {
      unsigned l = ldsa + j * 16u;
      unsigned g = goff + j * 16u;
      asm volatile("global_load_async_to_lds_b128 %0, %1, %2"
                   :: "v"(l), "v"(g), "s"(hsb)
                   : "memory");
    }
  }
  // Drain this wave's async copies, then barrier so all waves' tiles are ready.
  asm volatile("s_wait_asynccnt 0x0" ::: "memory");
  __syncthreads();

  const int wave = tid >> 5;
  const int lane = tid & 31;
  const int l16 = lane & 15;
  const int hi = lane >> 4;

  // ---------- Pass A: act^T = silu(Wg hs^T) * (Wu hs^T), per 16-wide i tile --
  for (int c = 0; c < I_DIM / 128; ++c) {
    const int i0 = c * 128 + wave * 16;
    v8f cg0 = {0,0,0,0,0,0,0,0}, cg1 = {0,0,0,0,0,0,0,0};
    v8f cu0 = {0,0,0,0,0,0,0,0}, cu1 = {0,0,0,0,0,0,0,0};
    const float* wgrow = wg + ((size_t)e * I_DIM + (i0 + l16)) * H_DIM;
    const float* wurow = wu + ((size_t)e * I_DIM + (i0 + l16)) * H_DIM;
    for (int k0 = 0; k0 < H_DIM; k0 += 32) {
      v16bf ag = load_a_f32(wgrow + k0 + 8 * hi);
      v16bf au = load_a_f32(wurow + k0 + 8 * hi);
      v16bf b0 = load_b_lds(hs_lds + (0  + l16) * HS_PITCH + k0 + 8 * hi);
      v16bf b1 = load_b_lds(hs_lds + (16 + l16) * HS_PITCH + k0 + 8 * hi);
      cg0 = wmma_bf16(ag, b0, cg0);
      cg1 = wmma_bf16(ag, b1, cg1);
      cu0 = wmma_bf16(au, b0, cu0);
      cu1 = wmma_bf16(au, b1, cu1);
    }
    // SiLU(g)*u -> bf16 into LDS, D layout: M=i -> r+8*hi, N=t -> l16.
    // sigmoid via v_rcp_f32 (1 ulp), avoids IEEE division expansion; error is
    // far below the bf16 quantization applied right after.
#pragma unroll
    for (int s = 0; s < 2; ++s) {
      v8f g = s ? cg1 : cg0;
      v8f u = s ? cu1 : cu0;
      __bf16* row = act_lds + (s * 16 + l16) * ACT_PITCH + i0 + 8 * hi;
#pragma unroll
      for (int r = 0; r < 8; r += 2) {
        float g0 = g[r], g1 = g[r + 1];
        float a0 = g0 * __builtin_amdgcn_rcpf(1.f + __expf(-g0)) * u[r];
        float a1 = g1 * __builtin_amdgcn_rcpf(1.f + __expf(-g1)) * u[r + 1];
        union { __bf16 h[2]; unsigned int w; } pk;
        pk.h[0] = (__bf16)a0; pk.h[1] = (__bf16)a1;
        *reinterpret_cast<unsigned int*>(row + r) = pk.w;
      }
    }
  }
  __syncthreads();

  // ---------- Pass B: y^T = Wd act^T; each wave owns 128 H rows --------------
  v8f y[8][2];
  v8f zero = {0,0,0,0,0,0,0,0};
#pragma unroll
  for (int a = 0; a < 8; ++a) { y[a][0] = zero; y[a][1] = zero; }
  const int hbase = wave * 128;
  for (int k0 = 0; k0 < I_DIM; k0 += 32) {
    v16bf b0 = load_b_lds(act_lds + (0  + l16) * ACT_PITCH + k0 + 8 * hi);
    v16bf b1 = load_b_lds(act_lds + (16 + l16) * ACT_PITCH + k0 + 8 * hi);
#pragma unroll
    for (int ht = 0; ht < 8; ++ht) {
      const float* wdrow =
          wd + ((size_t)e * H_DIM + hbase + ht * 16 + l16) * I_DIM + k0 + 8 * hi;
      v16bf a = load_a_f32(wdrow);
      y[ht][0] = wmma_bf16(a, b0, y[ht][0]);
      y[ht][1] = wmma_bf16(a, b1, y[ht][1]);
    }
  }

  // Store unweighted y rows to per-slot partial buffer (no atomics).
#pragma unroll
  for (int ht = 0; ht < 8; ++ht) {
#pragma unroll
    for (int s = 0; s < 2; ++s) {
      int t_loc = s * 16 + l16;
      if (t_loc < nvalid) {
        float* dst = part + (size_t)(base_slot + t_loc) * H_DIM +
                     hbase + ht * 16 + 8 * hi;
        union { v8f v; float4 q[2]; } o; o.v = y[ht][s];
        *reinterpret_cast<float4*>(dst) = o.q[0];
        *reinterpret_cast<float4*>(dst + 4) = o.q[1];
      }
    }
  }
}

// ---- kernel 6: combine out[t,h] = w0*part[s0,h] + w1*part[s1,h] -------------
__global__ __launch_bounds__(256) void k_combine(const float* __restrict__ part,
                                                 const int* __restrict__ slot_of,
                                                 const float* __restrict__ top_w,
                                                 float* __restrict__ out) {
  int idx = blockIdx.x * blockDim.x + threadIdx.x;
  int t = idx >> 10, h = idx & (H_DIM - 1);
  float w0 = top_w[t * 2], w1 = top_w[t * 2 + 1];
  int s0 = slot_of[t * 2], s1 = slot_of[t * 2 + 1];
  out[idx] = w0 * part[(size_t)s0 * H_DIM + h] +
             w1 * part[(size_t)s1 * H_DIM + h];
}

// ---- host launcher ----------------------------------------------------------
extern "C" void kernel_launch(void* const* d_in, const int* in_sizes, int n_in,
                              void* d_out, int out_size, void* d_ws, size_t ws_size,
                              hipStream_t stream) {
  const float* hs = (const float*)d_in[0];   // [T,H]
  const float* gw = (const float*)d_in[1];   // [E,H]
  const float* wg = (const float*)d_in[2];   // [E,I,H]
  const float* wu = (const float*)d_in[3];   // [E,I,H]
  const float* wd = (const float*)d_in[4];   // [E,H,I]
  float* out = (float*)d_out;

  char* ws = (char*)d_ws;
  __bf16* hsb   = (__bf16*)ws;                                   // 2 MB
  float*  part  = (float*)(ws + (size_t)2 * 1024 * 1024);        // 8 MB
  char*   meta  = ws + (size_t)10 * 1024 * 1024;
  int*   counts  = (int*)(meta + 0);
  int*   cursors = (int*)(meta + 64);
  int*   offsets = (int*)(meta + 128);
  int*   top_idx = (int*)(meta + 192);                           // 8 KB
  float* top_w   = (float*)(meta + 192 + 8192);                  // 8 KB
  int*   slot_of = (int*)(meta + 192 + 16384);                   // 8 KB
  int*   list_tok= (int*)(meta + 192 + 24576);                   // 8 KB

  k_init<<<1024, 256, 0, stream>>>(hs, hsb, counts, cursors);
  k_router<<<T_DIM / 16, 256, 0, stream>>>(hs, gw, counts, top_idx, top_w);
  k_scan<<<1, 32, 0, stream>>>(counts, offsets);
  k_fill<<<T_DIM / 256, 256, 0, stream>>>(top_idx, offsets, cursors,
                                          list_tok, slot_of);
  k_expert<<<dim3(MAX_SLOTS / TT, E_DIM), 256, 0, stream>>>(
      wg, wu, wd, hsb, counts, offsets, list_tok, part);
  k_combine<<<(T_DIM * H_DIM) / 256, 256, 0, stream>>>(part, slot_of, top_w, out);
}
